// GraphTransformerNet_74517682585745
// MI455X (gfx1250) — compile-verified, matrix operand order checked by reading, not run on hardware
//
#include <hip/hip_runtime.h>
#include <hip/hip_bf16.h>

typedef __attribute__((ext_vector_type(16))) _Float16 v16h;
typedef __attribute__((ext_vector_type(8)))  _Float16 v8h;
typedef __attribute__((ext_vector_type(8)))  float    v8f;
typedef __attribute__((ext_vector_type(4)))  _Float16 v4h;

// ---------------------------------------------------------------------------
// Utility kernels
// ---------------------------------------------------------------------------
__global__ void k_zero_f32(float* __restrict__ p, long n) {
  long i = (long)blockIdx.x * blockDim.x + threadIdx.x;
  if (i < n) p[i] = 0.0f;
}

// f32 -> f16, row-major, zero pad
__global__ void k_cast_pad(const float* __restrict__ src, _Float16* __restrict__ dst,
                           int srcRows, int srcCols, int dstRows, int dstCols) {
  long i = (long)blockIdx.x * blockDim.x + threadIdx.x;
  long total = (long)dstRows * dstCols;
  if (i >= total) return;
  int r = (int)(i / dstCols);
  int c = (int)(i % dstCols);
  float v = (r < srcRows && c < srcCols) ? src[(long)r * srcCols + c] : 0.0f;
  dst[i] = (_Float16)v;
}

// f32 [K,N] -> f16 transposed [dstN][dstK], zero pad
__global__ void k_cast_pad_t(const float* __restrict__ src, _Float16* __restrict__ dstT,
                             int srcK, int srcN, int dstK, int dstN) {
  long i = (long)blockIdx.x * blockDim.x + threadIdx.x;
  long total = (long)dstN * dstK;
  if (i >= total) return;
  int n = (int)(i / dstK);
  int k = (int)(i % dstK);
  float v = (k < srcK && n < srcN) ? src[(long)k * srcN + n] : 0.0f;
  dstT[i] = (_Float16)v;
}

// ---------------------------------------------------------------------------
// WMMA GEMM: C[M,N] = A[M,K] (f16 row-major) x Bt[N,K] (f16, B transposed).
// 128x64 macro-tile / 256 threads; 8 waves as 4(M) x 2(N); each wave owns a
// 32x32 patch = 4 x v_wmma_f32_16x16x32_f16 per 32-wide K-step.
// Double-buffered LDS ping-pong: the next tile's global_load_async_to_lds_b128
// group is issued before computing the current tile; s_wait_asynccnt 0x3 keeps
// the 3 new loads in flight (async loads complete in order), fully hiding
// global latency behind the WMMA stage.
// OOB handled by address clamping (OOB rows/cols only feed masked stores).
// Optional fused 3-way gather for the edge readout:
//   k<128: hh[src[row]]  k<256: ee[row]  else: hh[dst[row]]
// ---------------------------------------------------------------------------
__global__ __launch_bounds__(256) void k_wmma_gemm(
    const _Float16* __restrict__ A, int lda,
    const _Float16* __restrict__ Bt, int ldbt, int Nld,
    float* __restrict__ Cf, _Float16* __restrict__ Ch, int ldc,
    const float* __restrict__ bias, int relu,
    int M, int Nstore, int Kdim,
    const int* __restrict__ gsrc, const int* __restrict__ gdst,
    const _Float16* __restrict__ ghh, const _Float16* __restrict__ gee)
{
  __shared__ _Float16 As[2][128][40];  // 2 x (128 rows x 32 k (+8 pad)) : 20480 B
  __shared__ _Float16 Bs[2][64][40];   // 2 x (64 cols x 32 k (+8 pad))  : 10240 B

  const int tid   = threadIdx.x;
  const int lane  = tid & 31;
  const int wid   = tid >> 5;
  const int tileM = blockIdx.y * 128;
  const int tileN = blockIdx.x * 64;

  // A staging: 128 rows, 2 threads/row, 16 halfs (2 x b128) each
  const int ar  = tid >> 1;
  const int akc = (tid & 1) << 4;
  // B staging: 64 cols, 4 threads/col, 8 halfs (1 x b128) each
  const int bcl = tid >> 2;
  const int bkc = (tid & 3) << 3;

  // clamped source coordinates (OOB rows/cols only affect masked stores)
  const int arow = (tileM + ar < M) ? (tileM + ar) : (M - 1);
  const int bcol = (tileN + bcl < Nld) ? (tileN + bcl) : (Nld - 1);

  // hoist gather indices / base pointers out of the K-loop
  long g_src_off = 0, g_dst_off = 0;
  const _Float16* aBase = nullptr;
  if (gsrc) {
    g_src_off = (long)gsrc[arow] * 128;
    g_dst_off = (long)gdst[arow] * 128;
  } else {
    aBase = A + (long)arow * lda + akc;
  }
  const _Float16* bBase = Bt + (long)bcol * ldbt + bkc;

  // LDS destination byte addresses, per ping-pong buffer
  const unsigned ldsA[2] = { (unsigned)(uintptr_t)&As[0][ar][akc],
                             (unsigned)(uintptr_t)&As[1][ar][akc] };
  const unsigned ldsB[2] = { (unsigned)(uintptr_t)&Bs[0][bcl][bkc],
                             (unsigned)(uintptr_t)&Bs[1][bcl][bkc] };

  auto prefetch = [&](int kBase, int buf) {
    const _Float16* ap;
    if (gsrc) {
      int k0 = kBase + akc;
      if (k0 < 128)      ap = ghh + g_src_off + k0;
      else if (k0 < 256) ap = gee + (long)arow * 128 + (k0 - 128);
      else               ap = ghh + g_dst_off + (k0 - 256);
    } else {
      ap = aBase + kBase;
    }
    unsigned long long ga0 = (unsigned long long)(uintptr_t)ap;
    unsigned long long ga1 = ga0 + 16ull;
    unsigned long long gb  = (unsigned long long)(uintptr_t)(bBase + kBase);
    asm volatile(
        "global_load_async_to_lds_b128 %0, %2, off\n\t"
        "global_load_async_to_lds_b128 %1, %3, off"
        :: "v"(ldsA[buf]), "v"(ldsA[buf] + 16u), "v"(ga0), "v"(ga1) : "memory");
    asm volatile(
        "global_load_async_to_lds_b128 %0, %1, off"
        :: "v"(ldsB[buf]), "v"(gb) : "memory");
  };

  // wave tile coordinates
  const int wm = (wid >> 1) << 5;   // 0,32,64,96
  const int wn = (wid & 1) << 5;    // 0,32
  const int am = lane & 15;
  const int kb = (lane >> 4) << 3;  // 0 or 8

  v8f c00 = {}, c01 = {}, c10 = {}, c11 = {};

  const int steps = Kdim >> 5;
  prefetch(0, 0);
  for (int i = 0; i < steps; ++i) {
    const int cur = i & 1;
    if (i + 1 < steps) {
      prefetch((i + 1) << 5, cur ^ 1);            // overlaps with compute below
      asm volatile("s_wait_asynccnt 0x3" ::: "memory");  // current tile landed
    } else {
      asm volatile("s_wait_asynccnt 0x0" ::: "memory");
    }
    __syncthreads();   // all waves' current-tile data visible

    // ---- fragments: contiguous 16B LDS reads (ds_load_b128) ----
    v8h a0l = *(const v8h*)&As[cur][wm + am][kb];
    v8h a0h = *(const v8h*)&As[cur][wm + am][kb + 16];
    v8h a1l = *(const v8h*)&As[cur][wm + 16 + am][kb];
    v8h a1h = *(const v8h*)&As[cur][wm + 16 + am][kb + 16];
    v8h b0l = *(const v8h*)&Bs[cur][wn + am][kb];
    v8h b0h = *(const v8h*)&Bs[cur][wn + am][kb + 16];
    v8h b1l = *(const v8h*)&Bs[cur][wn + 16 + am][kb];
    v8h b1h = *(const v8h*)&Bs[cur][wn + 16 + am][kb + 16];
    v16h a0 = __builtin_shufflevector(a0l, a0h, 0,1,2,3,4,5,6,7,8,9,10,11,12,13,14,15);
    v16h a1 = __builtin_shufflevector(a1l, a1h, 0,1,2,3,4,5,6,7,8,9,10,11,12,13,14,15);
    v16h b0 = __builtin_shufflevector(b0l, b0h, 0,1,2,3,4,5,6,7,8,9,10,11,12,13,14,15);
    v16h b1 = __builtin_shufflevector(b1l, b1h, 0,1,2,3,4,5,6,7,8,9,10,11,12,13,14,15);

    c00 = __builtin_amdgcn_wmma_f32_16x16x32_f16(false, a0, false, b0, (short)0, c00, false, false);
    c01 = __builtin_amdgcn_wmma_f32_16x16x32_f16(false, a0, false, b1, (short)0, c01, false, false);
    c10 = __builtin_amdgcn_wmma_f32_16x16x32_f16(false, a1, false, b0, (short)0, c10, false, false);
    c11 = __builtin_amdgcn_wmma_f32_16x16x32_f16(false, a1, false, b1, (short)0, c11, false, false);

    __syncthreads();   // reads done before iteration i+1's prefetch overwrites this buffer
  }

  // ---- epilogue ----
  const int row0 = tileM + wm + ((lane >> 4) << 3);
  const int col0 = tileN + wn + am;
  const int col1 = col0 + 16;

  auto store_tile = [&](v8f acc, int rowBase, int col) {
    #pragma unroll
    for (int r = 0; r < 8; ++r) {
      int row = rowBase + r;
      if (row >= M || col >= Nstore) continue;
      float v = acc[r];
      if (bias) v += bias[col];
      if (relu) v = fmaxf(v, 0.0f);
      if (Ch) Ch[(long)row * ldc + col] = (_Float16)v;
      else    Cf[(long)row * ldc + col] = v;
    }
  };
  store_tile(c00, row0, col0);
  store_tile(c01, row0, col1);
  store_tile(c10, row0 + 16, col0);
  store_tile(c11, row0 + 16, col1);
}

// ---------------------------------------------------------------------------
// Edge attention: one wave per edge. score = K[src]*Q[dst]*(1/4)*pe,
// per-head exp(clamp(sum)), e_out (f16), segment-sum via fp32 atomics.
// ---------------------------------------------------------------------------
__global__ __launch_bounds__(256) void k_score(
    const float* __restrict__ Q, const float* __restrict__ K,
    const float* __restrict__ V, const float* __restrict__ pe,
    const int* __restrict__ src, const int* __restrict__ dst,
    _Float16* __restrict__ e_out, float* __restrict__ wV,
    float* __restrict__ z, int E)
{
  int e = blockIdx.x * 8 + (threadIdx.x >> 5);
  if (e >= E) return;
  int lane = threadIdx.x & 31;
  int si = src[e], di = dst[e];
  const float scale = 0.25f;  // 1/sqrt(16)

  float4 kk = *(const float4*)(K + (long)si * 128 + lane * 4);
  float4 qq = *(const float4*)(Q + (long)di * 128 + lane * 4);
  float4 pp = *(const float4*)(pe + (long)e * 128 + lane * 4);
  float s0 = kk.x * qq.x * scale * pp.x;
  float s1 = kk.y * qq.y * scale * pp.y;
  float s2 = kk.z * qq.z * scale * pp.z;
  float s3 = kk.w * qq.w * scale * pp.w;

  v4h ho;
  ho[0] = (_Float16)s0; ho[1] = (_Float16)s1;
  ho[2] = (_Float16)s2; ho[3] = (_Float16)s3;
  *(v4h*)(e_out + (long)e * 128 + lane * 4) = ho;

  // head = lane>>2 (4 lanes x 4 dims = 16 dims per head)
  float sum = s0 + s1 + s2 + s3;
  sum += __shfl_xor(sum, 1);
  sum += __shfl_xor(sum, 2);
  float sh = __expf(fminf(fmaxf(sum, -5.0f), 5.0f));

  float4 vv = *(const float4*)(V + (long)si * 128 + lane * 4);
  float* wp = wV + (long)di * 128 + lane * 4;
  unsafeAtomicAdd(wp + 0, vv.x * sh);
  unsafeAtomicAdd(wp + 1, vv.y * sh);
  unsafeAtomicAdd(wp + 2, vv.z * sh);
  unsafeAtomicAdd(wp + 3, vv.w * sh);
  if ((lane & 3) == 0) unsafeAtomicAdd(&z[(long)di * 8 + (lane >> 2)], sh);
}

// h_out = wV / (z + 1e-6), f16 for next GEMM
__global__ void k_hout(const float* __restrict__ wV, const float* __restrict__ z,
                       _Float16* __restrict__ out, long n) {
  long i = (long)blockIdx.x * blockDim.x + threadIdx.x;
  if (i >= n) return;
  long row = i >> 7;
  int col = (int)(i & 127);
  out[i] = (_Float16)(wV[i] / (z[row * 8 + (col >> 4)] + 1e-6f));
}

// Per-column sum / sumsq (128 cols) with fp32 atomics into stats[256]
__global__ __launch_bounds__(128) void k_colstats(const float* __restrict__ X,
                                                  int rows, float* __restrict__ stats) {
  int col = threadIdx.x;
  int r0 = blockIdx.x * 512;
  int r1 = r0 + 512; if (r1 > rows) r1 = rows;
  float s = 0.f, q = 0.f;
  for (int r = r0; r < r1; ++r) {
    float x = X[(long)r * 128 + col];
    s += x; q += x * x;
  }
  unsafeAtomicAdd(&stats[col], s);
  unsafeAtomicAdd(&stats[128 + col], q);
}

// training-mode BN apply, output f16
__global__ void k_bn(const float* __restrict__ X, const float* __restrict__ stats,
                     const float* __restrict__ g, const float* __restrict__ b,
                     _Float16* __restrict__ out, long rows) {
  long i = (long)blockIdx.x * blockDim.x + threadIdx.x;
  if (i >= rows * 128) return;
  int col = (int)(i & 127);
  float inv = 1.0f / (float)rows;
  float m = stats[col] * inv;
  float v = stats[128 + col] * inv - m * m;
  float y = (X[i] - m) * rsqrtf(v + 1e-5f) * g[col] + b[col];
  out[i] = (_Float16)y;
}

// ---------------------------------------------------------------------------
// Host orchestration
// ---------------------------------------------------------------------------
extern "C" void kernel_launch(void* const* d_in, const int* in_sizes, int n_in,
                              void* d_out, int out_size, void* d_ws, size_t ws_size,
                              hipStream_t stream) {
  const int N = 50000, E = 600000;

  const float* h   = (const float*)d_in[0];
  const float* ein = (const float*)d_in[1];
  const int*   src = (const int*)d_in[2];
  const int*   dst = (const int*)d_in[3];
  const float* Wq  = (const float*)d_in[4];
  const float* Wk  = (const float*)d_in[5];
  const float* Wv  = (const float*)d_in[6];
  const float* Wpe = (const float*)d_in[7];
  const float* Woh = (const float*)d_in[8];
  const float* boh = (const float*)d_in[9];
  const float* Woe = (const float*)d_in[10];
  const float* boe = (const float*)d_in[11];
  const float* g1h = (const float*)d_in[12];
  const float* b1h = (const float*)d_in[13];
  const float* g1e = (const float*)d_in[14];
  const float* b1e = (const float*)d_in[15];
  const float* W1  = (const float*)d_in[16];
  const float* b1  = (const float*)d_in[17];
  const float* W2  = (const float*)d_in[18];
  const float* b2  = (const float*)d_in[19];

  char* ws = (char*)d_ws;
  size_t off = 0;
  auto alloc = [&](size_t bytes) -> char* {
    char* p = ws + off;
    off = (off + bytes + 255) & ~(size_t)255;
    return p;
  };

  // transposed f16 weights: [N][K] layout
  _Float16* WqT  = (_Float16*)alloc(128 * 128 * 2);
  _Float16* WkT  = (_Float16*)alloc(128 * 128 * 2);
  _Float16* WvT  = (_Float16*)alloc(128 * 128 * 2);
  _Float16* WohT = (_Float16*)alloc(128 * 128 * 2);
  _Float16* WoeT = (_Float16*)alloc(128 * 128 * 2);
  _Float16* WpeT = (_Float16*)alloc(128 * 160 * 2);   // K padded 130->160
  _Float16* W1T  = (_Float16*)alloc(192 * 384 * 2);
  _Float16* W2T  = (_Float16*)alloc(16 * 192 * 2);    // N padded 4->16
  _Float16* h16  = (_Float16*)alloc((size_t)N * 128 * 2);
  _Float16* e16  = (_Float16*)alloc((size_t)E * 160 * 2);  // reused as e_out16
  float*    Qf   = (float*)alloc((size_t)N * 128 * 4);
  float*    Kf   = (float*)alloc((size_t)N * 128 * 4);
  float*    Vf   = (float*)alloc((size_t)N * 128 * 4);
  float*    pef  = (float*)alloc((size_t)E * 128 * 4);     // reused as ee_pre
  float*    wV   = (float*)alloc((size_t)N * 128 * 4);     // zero region start
  float*    zf   = (float*)alloc((size_t)N * 8 * 4);
  float*    statsH = (float*)alloc(256 * 4);
  float*    statsE = (float*)alloc(256 * 4);
  _Float16* hout16 = (_Float16*)alloc((size_t)N * 128 * 2);
  float*    hh_pre = (float*)alloc((size_t)N * 128 * 4);
  _Float16* hh16   = (_Float16*)alloc((size_t)N * 128 * 2);
  _Float16* ee16   = (_Float16*)alloc((size_t)E * 128 * 2);
  _Float16* y16    = (_Float16*)alloc((size_t)E * 192 * 2);
  (void)ws_size; (void)in_sizes; (void)n_in; (void)out_size;

  _Float16* e_out16 = e16;   // alias: e16 consumed by pe GEMM before score writes
  float*    ee_pre  = pef;   // alias: pe consumed by score before Woe GEMM writes

  auto cdiv = [](long a, long b) { return (int)((a + b - 1) / b); };

  // zero wV + z + statsH + statsE (contiguous: all sizes are 256B multiples)
  long ztot = (long)N * 128 + (long)N * 8 + 512;
  k_zero_f32<<<cdiv(ztot, 256), 256, 0, stream>>>(wV, ztot);

  // weight casts (transposed) + activation casts (row-major)
  k_cast_pad_t<<<cdiv(128 * 128, 256), 256, 0, stream>>>(Wq, WqT, 128, 128, 128, 128);
  k_cast_pad_t<<<cdiv(128 * 128, 256), 256, 0, stream>>>(Wk, WkT, 128, 128, 128, 128);
  k_cast_pad_t<<<cdiv(128 * 128, 256), 256, 0, stream>>>(Wv, WvT, 128, 128, 128, 128);
  k_cast_pad_t<<<cdiv(128 * 128, 256), 256, 0, stream>>>(Woh, WohT, 128, 128, 128, 128);
  k_cast_pad_t<<<cdiv(128 * 128, 256), 256, 0, stream>>>(Woe, WoeT, 128, 128, 128, 128);
  k_cast_pad_t<<<cdiv(128 * 160, 256), 256, 0, stream>>>(Wpe, WpeT, 130, 128, 160, 128);
  k_cast_pad_t<<<cdiv(192 * 384, 256), 256, 0, stream>>>(W1, W1T, 384, 192, 384, 192);
  k_cast_pad_t<<<cdiv(16 * 192, 256), 256, 0, stream>>>(W2, W2T, 192, 4, 192, 16);
  k_cast_pad<<<cdiv((long)N * 128, 256), 256, 0, stream>>>(h, h16, N, 128, N, 128);
  k_cast_pad<<<cdiv((long)E * 160, 256), 256, 0, stream>>>(ein, e16, E, 130, E, 160);

  dim3 blk(256);
  auto gemm = [&](const _Float16* A, int lda, const _Float16* Bt, int ldbt, int Nld,
                  float* Cf, _Float16* Ch, int ldc, const float* bias, int relu,
                  int M, int Ns, int Kd,
                  const int* gs, const int* gd,
                  const _Float16* ghh, const _Float16* gee) {
    dim3 grid((Nld + 63) / 64, (M + 127) / 128);
    k_wmma_gemm<<<grid, blk, 0, stream>>>(A, lda, Bt, ldbt, Nld, Cf, Ch, ldc,
                                          bias, relu, M, Ns, Kd, gs, gd, ghh, gee);
  };

  // Q/K/V projections
  gemm(h16, 128, WqT, 128, 128, Qf, nullptr, 128, nullptr, 0, N, 128, 128,
       nullptr, nullptr, nullptr, nullptr);
  gemm(h16, 128, WkT, 128, 128, Kf, nullptr, 128, nullptr, 0, N, 128, 128,
       nullptr, nullptr, nullptr, nullptr);
  gemm(h16, 128, WvT, 128, 128, Vf, nullptr, 128, nullptr, 0, N, 128, 128,
       nullptr, nullptr, nullptr, nullptr);
  // edge projection (K padded 130->160)
  gemm(e16, 160, WpeT, 160, 128, pef, nullptr, 128, nullptr, 0, E, 128, 160,
       nullptr, nullptr, nullptr, nullptr);

  // edge attention + segment sums
  k_score<<<cdiv(E, 8), 256, 0, stream>>>(Qf, Kf, Vf, pef, src, dst,
                                          e_out16, wV, zf, E);
  k_hout<<<cdiv((long)N * 128, 256), 256, 0, stream>>>(wV, zf, hout16, (long)N * 128);

  // O projections (+bias)
  gemm(hout16, 128, WohT, 128, 128, hh_pre, nullptr, 128, boh, 0, N, 128, 128,
       nullptr, nullptr, nullptr, nullptr);
  gemm(e_out16, 128, WoeT, 128, 128, ee_pre, nullptr, 128, boe, 0, E, 128, 128,
       nullptr, nullptr, nullptr, nullptr);

  // batchnorm (stats + apply, f16 out)
  k_colstats<<<cdiv(N, 512), 128, 0, stream>>>(hh_pre, N, statsH);
  k_colstats<<<cdiv(E, 512), 128, 0, stream>>>(ee_pre, E, statsE);
  k_bn<<<cdiv((long)N * 128, 256), 256, 0, stream>>>(hh_pre, statsH, g1h, b1h, hh16, N);
  k_bn<<<cdiv((long)E * 128, 256), 256, 0, stream>>>(ee_pre, statsE, g1e, b1e, ee16, E);

  // FFN1: fused gather [hh[src], ee, hh[dst]] @ W1 + b1, relu, f16 out
  gemm(nullptr, 0, W1T, 384, 192, nullptr, y16, 192, b1, 1, E, 192, 384,
       src, dst, hh16, ee16);
  // FFN2: y @ W2 (padded N=16) + b2 -> d_out [E,4] f32
  gemm(y16, 192, W2T, 192, 16, (float*)d_out, nullptr, 4, b2, 0, E, 4, 192,
       nullptr, nullptr, nullptr, nullptr);
}
